// RIComplexMultiheadAttention_42855183679941
// MI455X (gfx1250) — compile-verified, matrix operand order checked by reading, not run on hardware
//
#include <hip/hip_runtime.h>

// ---------------------------------------------------------------------------
// RI complex multi-head attention, MI455X (gfx1250, wave32, WMMA bf16).
// T=1024 B=8 E=512 H=8 D=64.  All matmuls via v_wmma_f32_16x16x32_bf16.
// ---------------------------------------------------------------------------

typedef __attribute__((ext_vector_type(16))) __bf16 v16bf;
typedef __attribute__((ext_vector_type(8)))  float  v8f;

namespace {

constexpr int kT  = 1024;
constexpr int kB  = 8;
constexpr int kE  = 512;
constexpr int kH  = 8;
constexpr int kD  = 64;
constexpr int kE3 = 3 * kE;        // 1536
constexpr int kM  = kT * kB;       // 8192 GEMM rows
constexpr float kScale = 0.125f;   // D^-0.5

// K-index pattern inside a 16-bit WMMA operand fragment (ISA 7.12.2):
// VGPR v in 0..3 holds K = 2v,2v+1 (+8 for lanes 16-31); v in 4..7 holds K=16+...
__device__ __forceinline__ int kpat(int v, int half) {
  return (v < 4) ? (2 * v + 8 * half) : (16 + 2 * (v - 4) + 8 * half);
}

// A fragment: A[m][k], memory row-major [m][k], lane%16 selects m.
__device__ __forceinline__ v16bf load_a_bf16(const __bf16* p, size_t ld, int m0, int k0) {
  const int lane = threadIdx.x & 31;
  const int half = lane >> 4;
  const __bf16* row = p + (size_t)(m0 + (lane & 15)) * ld + k0;
  v16bf f;
#pragma unroll
  for (int v = 0; v < 8; ++v) {
    const int kb = kpat(v, half);
    f[2 * v]     = row[kb];
    f[2 * v + 1] = row[kb + 1];
  }
  return f;
}

// A fragment from f32 source (softmax probabilities), converted to bf16.
__device__ __forceinline__ v16bf load_a_f32(const float* p, size_t ld, int m0, int k0) {
  const int lane = threadIdx.x & 31;
  const int half = lane >> 4;
  const float* row = p + (size_t)(m0 + (lane & 15)) * ld + k0;
  v16bf f;
#pragma unroll
  for (int v = 0; v < 8; ++v) {
    const int kb = kpat(v, half);
    f[2 * v]     = (__bf16)row[kb];
    f[2 * v + 1] = (__bf16)row[kb + 1];
  }
  return f;
}

// B fragment of B[k][n] when memory holds element(row=n, col=k)  (W[n][k], K[s][d]).
__device__ __forceinline__ v16bf load_b_trans(const __bf16* p, size_t ld, int n0, int k0) {
  return load_a_bf16(p, ld, n0, k0);  // same gather pattern, roles swapped
}

// B fragment of B[k][n] when memory holds element(row=k, col=n)  (V[s][d]).
__device__ __forceinline__ v16bf load_b_direct(const __bf16* p, size_t ld, int k0, int n0) {
  const int lane = threadIdx.x & 31;
  const int half = lane >> 4;
  const int n = n0 + (lane & 15);
  v16bf f;
#pragma unroll
  for (int v = 0; v < 8; ++v) {
    const int kb = kpat(v, half);
    f[2 * v]     = p[(size_t)(k0 + kb) * ld + n];
    f[2 * v + 1] = p[(size_t)(k0 + kb + 1) * ld + n];
  }
  return f;
}

__device__ __forceinline__ v8f wmma_bf16(v16bf a, v16bf b, v8f c) {
  return __builtin_amdgcn_wmma_f32_16x16x32_bf16(false, a, false, b, (short)0, c,
                                                 false, false);
}

}  // namespace

// ---------------------------------------------------------------------------
// elementwise helpers (float4-vectorized: pure bandwidth, ~120MB touched)
// ---------------------------------------------------------------------------
__global__ void cast_f32_bf16_kernel(const float4* __restrict__ in,
                                     __bf16* __restrict__ out, int n4) {
  const int i = blockIdx.x * blockDim.x + threadIdx.x;
  if (i < n4) {
    const float4 v = in[i];
    __bf16 o[4] = {(__bf16)v.x, (__bf16)v.y, (__bf16)v.z, (__bf16)v.w};
    *(uint2*)(out + 4 * (size_t)i) = *(const uint2*)o;
  }
}

__global__ void zero_f32_kernel(float4* __restrict__ p, int n4) {
  const int i = blockIdx.x * blockDim.x + threadIdx.x;
  if (i < n4) p[i] = make_float4(0.f, 0.f, 0.f, 0.f);
}

// ---------------------------------------------------------------------------
// Complex linear: y_r = Xr Wr^T - Xi Wi^T + br ; y_i = Xi Wr^T + Xr Wi^T + bi
// Block tile 128(M) x 64(N), 8 waves as 4x2, wave tile 32x32 (2x2 WMMA tiles),
// K-step 32.  W stored [N][K] (PyTorch Linear layout).  LDS double-buffered,
// staged with b128 transactions (rows are 64B-aligned at every K-step).
// ---------------------------------------------------------------------------
__device__ __forceinline__ void stage_panels(
    const __bf16* __restrict__ Xr, const __bf16* __restrict__ Xi,
    const __bf16* __restrict__ Wr, const __bf16* __restrict__ Wi,
    uint4* __restrict__ dXr, uint4* __restrict__ dXi,
    uint4* __restrict__ dWr, uint4* __restrict__ dWi,
    int m0, int n0, int k0, int K, int tid) {
  // X panels: 128 rows x 32 bf16 = 512 uint4 each; 2 per thread per array.
#pragma unroll
  for (int t = 0; t < 2; ++t) {
    const int i = tid + t * 256;
    const int r = i >> 2, c = i & 3;
    dXr[i] = ((const uint4*)(Xr + (size_t)(m0 + r) * K + k0))[c];
    dXi[i] = ((const uint4*)(Xi + (size_t)(m0 + r) * K + k0))[c];
  }
  // W panels: 64 rows x 32 bf16 = 256 uint4 each; 1 per thread per array.
  {
    const int r = tid >> 2, c = tid & 3;
    dWr[tid] = ((const uint4*)(Wr + (size_t)(n0 + r) * K + k0))[c];
    dWi[tid] = ((const uint4*)(Wi + (size_t)(n0 + r) * K + k0))[c];
  }
}

template <bool WRITE_BF16>
__global__ __launch_bounds__(256) void cplx_gemm_kernel(
    const __bf16* __restrict__ Xr, const __bf16* __restrict__ Xi,
    const __bf16* __restrict__ Wr, const __bf16* __restrict__ Wi,
    const float* __restrict__ br, const float* __restrict__ bi,
    __bf16* __restrict__ Yr16, __bf16* __restrict__ Yi16,
    float* __restrict__ Yr32, float* __restrict__ Yi32,
    const int N, const int K) {
  __shared__ uint4 sXr[2][512];  // 128x32 bf16, double-buffered
  __shared__ uint4 sXi[2][512];
  __shared__ uint4 sWr[2][256];  // 64x32 bf16
  __shared__ uint4 sWi[2][256];

  const int tid  = threadIdx.x;
  const int lane = tid & 31;
  const int w    = tid >> 5;
  const int wm   = w >> 1;  // 0..3
  const int wn   = w & 1;   // 0..1
  const int m0   = blockIdx.x * 128;
  const int n0   = blockIdx.y * 64;

  const v8f vzero = {};
  v8f accR[2][2], accN[2][2], accI[2][2];
#pragma unroll
  for (int mt = 0; mt < 2; ++mt)
#pragma unroll
    for (int nt = 0; nt < 2; ++nt) {
      accR[mt][nt] = vzero;
      accN[mt][nt] = vzero;
      accI[mt][nt] = vzero;
    }

  stage_panels(Xr, Xi, Wr, Wi, sXr[0], sXi[0], sWr[0], sWi[0], m0, n0, 0, K, tid);
  __syncthreads();

  for (int k0 = 0; k0 < K; k0 += 32) {
    const int cur = (k0 >> 5) & 1;
    if (k0 + 32 < K)  // prefetch next panel into the other buffer
      stage_panels(Xr, Xi, Wr, Wi, sXr[cur ^ 1], sXi[cur ^ 1], sWr[cur ^ 1],
                   sWi[cur ^ 1], m0, n0, k0 + 32, K, tid);
    if (k0 + 64 < K) {  // global_prefetch of the panel after that
      __builtin_prefetch(&Xr[(size_t)(m0 + (tid >> 1)) * K + (k0 + 64)], 0, 1);
      __builtin_prefetch(&Xi[(size_t)(m0 + (tid >> 1)) * K + (k0 + 64)], 0, 1);
    }

    const __bf16* pXr = (const __bf16*)sXr[cur];
    const __bf16* pXi = (const __bf16*)sXi[cur];
    const __bf16* pWr = (const __bf16*)sWr[cur];
    const __bf16* pWi = (const __bf16*)sWi[cur];

    v16bf axr[2], axi[2], bwr[2], bwi[2];
#pragma unroll
    for (int mt = 0; mt < 2; ++mt) {
      axr[mt] = load_a_bf16(pXr, 32, wm * 32 + mt * 16, 0);
      axi[mt] = load_a_bf16(pXi, 32, wm * 32 + mt * 16, 0);
    }
#pragma unroll
    for (int nt = 0; nt < 2; ++nt) {
      bwr[nt] = load_b_trans(pWr, 32, wn * 32 + nt * 16, 0);
      bwi[nt] = load_b_trans(pWi, 32, wn * 32 + nt * 16, 0);
    }
#pragma unroll
    for (int mt = 0; mt < 2; ++mt)
#pragma unroll
      for (int nt = 0; nt < 2; ++nt) {
        accR[mt][nt] = wmma_bf16(axr[mt], bwr[nt], accR[mt][nt]);  // xr*Wr
        accN[mt][nt] = wmma_bf16(axi[mt], bwi[nt], accN[mt][nt]);  // xi*Wi
        accI[mt][nt] = wmma_bf16(axi[mt], bwr[nt], accI[mt][nt]);  // xi*Wr
        accI[mt][nt] = wmma_bf16(axr[mt], bwi[nt], accI[mt][nt]);  // +xr*Wi
      }
    __syncthreads();  // one barrier per K-step (double buffer)
  }

  const int half = lane >> 4;
#pragma unroll
  for (int mt = 0; mt < 2; ++mt)
#pragma unroll
    for (int nt = 0; nt < 2; ++nt) {
      const int n = n0 + wn * 32 + nt * 16 + (lane & 15);
      const float bre = br[n];
      const float bim = bi[n];
#pragma unroll
      for (int vg = 0; vg < 8; ++vg) {
        const int m = m0 + wm * 32 + mt * 16 + vg + half * 8;
        const float yr = accR[mt][nt][vg] - accN[mt][nt][vg] + bre;
        const float yi = accI[mt][nt][vg] + bim;
        const size_t idx = (size_t)m * N + n;
        if (WRITE_BF16) {
          Yr16[idx] = (__bf16)yr;
          Yi16[idx] = (__bf16)yi;
        } else {
          Yr32[idx] = yr;
          Yi32[idx] = yi;
        }
      }
    }
}

// ---------------------------------------------------------------------------
// Fused complex attention.  One block = one (b,h) head x 16 query rows.
// Score panel (16 x 1024 f32, real+imag = 128KB) lives entirely in LDS,
// enabled by CDNA5's 320KB/WGP.  Phase1: QK^T via WMMA.  Phase2: wave32
// shfl softmax + atomic head-mean into d_out.  Phase3: P.V via WMMA with
// the K(=s) dimension split over the 8 waves, LDS cross-wave reduction.
// ---------------------------------------------------------------------------
__global__ __launch_bounds__(256) void cplx_attn_kernel(
    const __bf16* __restrict__ qkvR, const __bf16* __restrict__ qkvI,
    __bf16* __restrict__ attnR, __bf16* __restrict__ attnI,
    float* __restrict__ awR, float* __restrict__ awI) {
  extern __shared__ char smem[];
  float* sR  = (float*)smem;              // [16][1024]
  float* sI  = sR + 16 * kT;              // [16][1024]
  __bf16* sQr = (__bf16*)(sI + 16 * kT);  // [16][64]
  __bf16* sQi = sQr + 16 * kD;
  float* red = sR;  // reused as reduction scratch (16384 floats)

  const int tid  = threadIdx.x;
  const int lane = tid & 31;
  const int w    = tid >> 5;
  const int t0   = blockIdx.x * 16;
  const int bh   = blockIdx.y;
  const int b    = bh >> 3;
  const int h    = bh & 7;
  const size_t ldRow = (size_t)kB * kE3;  // stride between consecutive t (or s)

  const __bf16* qR = qkvR + (size_t)b * kE3 + h * kD;
  const __bf16* qI = qkvI + (size_t)b * kE3 + h * kD;
  const __bf16* kR = qR + kE;
  const __bf16* kI = qI + kE;
  const __bf16* vR = qR + 2 * kE;
  const __bf16* vI = qI + 2 * kE;

  // Stage Q rows (16 x 64, r & i) into LDS via b128 (rows are 128B contiguous).
  if (tid < 128) {
    const int r = tid >> 3, c = tid & 7;
    ((uint4*)sQr)[tid] = ((const uint4*)(qR + (size_t)(t0 + r) * ldRow))[c];
    ((uint4*)sQi)[tid] = ((const uint4*)(qI + (size_t)(t0 + r) * ldRow))[c];
  }
  __syncthreads();

  const v8f vzero = {};
  const int half = lane >> 4;

  // ---- Phase 1: scores.  s_r = (qr.kr + qi.ki)*S ; s_i = (qi.kr - qr.ki)*S
  v16bf aqr[2], aqi[2];
#pragma unroll
  for (int ks = 0; ks < 2; ++ks) {
    aqr[ks] = load_a_bf16(sQr, kD, 0, ks * 32);
    aqi[ks] = load_a_bf16(sQi, kD, 0, ks * 32);
  }
  for (int j = 0; j < 8; ++j) {           // 8 key tiles of 16 per wave
    const int s0 = (w * 8 + j) * 16;
    v8f aR = vzero, aP = vzero, aN = vzero;
#pragma unroll
    for (int ks = 0; ks < 2; ++ks) {      // D = 64 = 2 K-steps
      const v16bf bkr = load_b_trans(kR, ldRow, s0, ks * 32);
      const v16bf bki = load_b_trans(kI, ldRow, s0, ks * 32);
      aR = wmma_bf16(aqr[ks], bkr, aR);
      aR = wmma_bf16(aqi[ks], bki, aR);
      aP = wmma_bf16(aqi[ks], bkr, aP);
      aN = wmma_bf16(aqr[ks], bki, aN);
    }
#pragma unroll
    for (int vg = 0; vg < 8; ++vg) {
      const int m = vg + half * 8;
      const int n = s0 + (lane & 15);
      sR[m * kT + n] = aR[vg] * kScale;
      sI[m * kT + n] = (aP[vg] - aN[vg]) * kScale;
    }
  }
  __syncthreads();

  // ---- Phase 2: softmax per row (2 rows per wave) + head-mean aw output.
  for (int rr = 0; rr < 2; ++rr) {
    const int row = w * 2 + rr;
#pragma unroll
    for (int arr = 0; arr < 2; ++arr) {
      float* srow = (arr ? sI : sR) + row * kT;
      float* awRow =
          (arr ? awI : awR) + ((size_t)b * kT + (t0 + row)) * (size_t)kT;
      float mx = -3.0e38f;
      for (int c = lane; c < kT; c += 32) mx = fmaxf(mx, srow[c]);
#pragma unroll
      for (int off = 16; off; off >>= 1) mx = fmaxf(mx, __shfl_xor(mx, off, 32));
      float sum = 0.f;
      for (int c = lane; c < kT; c += 32) {
        const float e = __expf(srow[c] - mx);
        srow[c] = e;
        sum += e;
      }
#pragma unroll
      for (int off = 16; off; off >>= 1) sum += __shfl_xor(sum, off, 32);
      const float inv = 1.0f / sum;
      for (int c = lane; c < kT; c += 32) {
        const float pv = srow[c] * inv;
        srow[c] = pv;
        atomicAdd(&awRow[c], pv * 0.125f);  // mean over H=8 heads
      }
    }
  }
  __syncthreads();

  // ---- Phase 3: attn_r = P_r.Vr + P_i.Vi ; attn_i = P_i.Vr - P_r.Vi
  // K(=s) dim split across waves: wave w owns s in [w*128, w*128+128).
  v8f pr[4], pp[4], pn[4];
#pragma unroll
  for (int nt = 0; nt < 4; ++nt) { pr[nt] = vzero; pp[nt] = vzero; pn[nt] = vzero; }
  const int kbase = w * 128;
  for (int kc = 0; kc < 4; ++kc) {
    const int k0 = kbase + kc * 32;
    const v16bf pA_r = load_a_f32(sR, kT, 0, k0);
    const v16bf pA_i = load_a_f32(sI, kT, 0, k0);
#pragma unroll
    for (int nt = 0; nt < 4; ++nt) {  // D = 64 = 4 N tiles
      const v16bf bvr = load_b_direct(vR, ldRow, k0, nt * 16);
      const v16bf bvi = load_b_direct(vI, ldRow, k0, nt * 16);
      pr[nt] = wmma_bf16(pA_r, bvr, pr[nt]);
      pr[nt] = wmma_bf16(pA_i, bvi, pr[nt]);
      pp[nt] = wmma_bf16(pA_i, bvr, pp[nt]);
      pn[nt] = wmma_bf16(pA_r, bvi, pn[nt]);
    }
  }
  __syncthreads();  // all waves done reading score panel; reuse sR as scratch
#pragma unroll
  for (int nt = 0; nt < 4; ++nt)
#pragma unroll
    for (int vg = 0; vg < 8; ++vg) {
      red[(((w * 4 + nt) * 2 + 0) * 256) + vg * 32 + lane] = pr[nt][vg];
      red[(((w * 4 + nt) * 2 + 1) * 256) + vg * 32 + lane] =
          pp[nt][vg] - pn[nt][vg];
    }
  __syncthreads();
  // Cross-wave reduce 2048 output values (16 rows x 64 dims x {r,i}).
  for (int s = tid; s < 2048; s += 256) {
    const int slot = s & 255;
    const int ri   = (s >> 8) & 1;
    const int nt   = s >> 9;
    float acc = 0.f;
#pragma unroll
    for (int ww = 0; ww < 8; ++ww)
      acc += red[(((ww * 4 + nt) * 2 + ri) * 256) + slot];
    const int vg = slot >> 5, ln = slot & 31;
    const int m = vg + ((ln >> 4) << 3);
    const int n = nt * 16 + (ln & 15);
    const size_t idx = ((size_t)(t0 + m) * kB + b) * kE + (h * kD + n);
    if (ri == 0) attnR[idx] = (__bf16)acc;
    else         attnI[idx] = (__bf16)acc;
  }
}

// ---------------------------------------------------------------------------
// host launcher
// ---------------------------------------------------------------------------
extern "C" void kernel_launch(void* const* d_in, const int* in_sizes, int n_in,
                              void* d_out, int out_size, void* d_ws,
                              size_t ws_size, hipStream_t stream) {
  const float* x_r  = (const float*)d_in[0];
  const float* x_i  = (const float*)d_in[1];
  const float* Wq_r = (const float*)d_in[2];
  const float* Wq_i = (const float*)d_in[3];
  const float* bq_r = (const float*)d_in[4];
  const float* bq_i = (const float*)d_in[5];
  const float* Wo_r = (const float*)d_in[6];
  const float* Wo_i = (const float*)d_in[7];
  const float* bo_r = (const float*)d_in[8];
  const float* bo_i = (const float*)d_in[9];
  float* out = (float*)d_out;

  // d_out layout: out_r[T,B,E] | out_i[T,B,E] | aw_r[B,T,T] | aw_i[B,T,T]
  const size_t nOut = (size_t)kM * kE;       // 4,194,304
  const size_t nAw  = (size_t)kB * kT * kT;  // 8,388,608
  float* out_r = out;
  float* out_i = out + nOut;
  float* aw_r  = out + 2 * nOut;
  float* aw_i  = aw_r + nAw;

  // workspace carve-out (all bf16 intermediates), ~84 MB total
  size_t off = 0;
  char* ws = (char*)d_ws;
  auto alloc = [&](size_t bytes) -> char* {
    char* p = ws + off;
    off += (bytes + 255) & ~((size_t)255);
    return p;
  };
  __bf16* xr16  = (__bf16*)alloc((size_t)kM * kE * 2);
  __bf16* xi16  = (__bf16*)alloc((size_t)kM * kE * 2);
  __bf16* wqr16 = (__bf16*)alloc((size_t)kE3 * kE * 2);
  __bf16* wqi16 = (__bf16*)alloc((size_t)kE3 * kE * 2);
  __bf16* wor16 = (__bf16*)alloc((size_t)kE * kE * 2);
  __bf16* woi16 = (__bf16*)alloc((size_t)kE * kE * 2);
  __bf16* qkvR  = (__bf16*)alloc((size_t)kM * kE3 * 2);
  __bf16* qkvI  = (__bf16*)alloc((size_t)kM * kE3 * 2);
  __bf16* attnR = (__bf16*)alloc((size_t)kM * kE * 2);
  __bf16* attnI = (__bf16*)alloc((size_t)kM * kE * 2);

  auto nblk4 = [](size_t n) { return (unsigned)((n / 4 + 255) / 256); };

  // fp32 -> bf16 casts (float4 vectorized; all sizes divisible by 4)
  cast_f32_bf16_kernel<<<nblk4(kM * kE), 256, 0, stream>>>(
      (const float4*)x_r, xr16, kM * kE / 4);
  cast_f32_bf16_kernel<<<nblk4(kM * kE), 256, 0, stream>>>(
      (const float4*)x_i, xi16, kM * kE / 4);
  cast_f32_bf16_kernel<<<nblk4(kE3 * kE), 256, 0, stream>>>(
      (const float4*)Wq_r, wqr16, kE3 * kE / 4);
  cast_f32_bf16_kernel<<<nblk4(kE3 * kE), 256, 0, stream>>>(
      (const float4*)Wq_i, wqi16, kE3 * kE / 4);
  cast_f32_bf16_kernel<<<nblk4(kE * kE), 256, 0, stream>>>(
      (const float4*)Wo_r, wor16, kE * kE / 4);
  cast_f32_bf16_kernel<<<nblk4(kE * kE), 256, 0, stream>>>(
      (const float4*)Wo_i, woi16, kE * kE / 4);

  // zero the attention-weight output region (atomic head-mean accumulation)
  zero_f32_kernel<<<nblk4(2 * nAw), 256, 0, stream>>>((float4*)aw_r,
                                                      (int)(2 * nAw / 4));

  // QKV complex linear: [8192 x 512] x [1536 x 512]^T -> bf16 qkv
  cplx_gemm_kernel<true><<<dim3(kM / 128, kE3 / 64), 256, 0, stream>>>(
      xr16, xi16, wqr16, wqi16, bq_r, bq_i, qkvR, qkvI, nullptr, nullptr, kE3, kE);

  // Fused attention: 64 row-blocks x 64 heads, 132KB LDS score panel per block
  const size_t attnLds = (size_t)2 * 16 * kT * sizeof(float)    // score panels
                       + (size_t)2 * 16 * kD * sizeof(__bf16);  // Q staging
  cplx_attn_kernel<<<dim3(kT / 16, kB * kH), 256, attnLds, stream>>>(
      qkvR, qkvI, attnR, attnI, aw_r, aw_i);

  // Output complex linear: [8192 x 512] x [512 x 512]^T -> f32 straight to d_out
  cplx_gemm_kernel<false><<<dim3(kM / 128, kE / 64), 256, 0, stream>>>(
      attnR, attnI, wor16, woi16, bo_r, bo_i, nullptr, nullptr, out_r, out_i, kE, kE);
}
